// MultiHeadAttention_35613868819134
// MI455X (gfx1250) — matrix operand order repaired and checked by reading, not compiled, not run on hardware
//
#include <hip/hip_runtime.h>
#include <hip/hip_bf16.h>
#include <math.h>
#include <stdint.h>

// ---------------------------------------------------------------------------
// MHA for MI455X (gfx1250): bf16 WMMA, f32 accumulation, async-LDS pipeline.
//   x:[4096,1024]f32, W*:[1024,1024]f32 (y = x @ W^T + b), 16 heads, d=64.
// Attention computes S^T = K*Q^T so the softmax output is directly the
// A-operand of P*V (no transpose / LDS round trip). K/V chunks are staged
// with global_load_async_to_lds_b128, double-buffered (ASYNCcnt).
// ---------------------------------------------------------------------------

#define BSZ     2
#define QLEN    2048
#define DIM     1024
#define NHEADS  16
#define DHEAD   64
#define MROWS   (BSZ * QLEN)   // 4096

typedef __attribute__((ext_vector_type(16))) __bf16 v16bf;
typedef __attribute__((ext_vector_type(8)))  __bf16 v8bf;
typedef __attribute__((ext_vector_type(8)))  float  v8f;

union V16 { v16bf v; v8bf h[2]; __bf16 e[16]; };

// Async copy 16B global -> LDS (ASYNCcnt-tracked).
__device__ __forceinline__ void async_b128_to_lds(void* lds, const void* gptr) {
    uint32_t lo = (uint32_t)(uintptr_t)lds;   // generic low 32 bits = LDS offset
    asm volatile("global_load_async_to_lds_b128 %0, %1, off"
                 :: "v"(lo), "v"(gptr) : "memory");
}

// ----------------------------- converters ----------------------------------

__global__ void cvt_f32_bf16(const float* __restrict__ in,
                             __bf16* __restrict__ out, int n) {
    int i = blockIdx.x * blockDim.x + threadIdx.x;
    if (i < n) out[i] = (__bf16)in[i];
}

// out[c*R + r] = (bf16) in[r*C + c]   (store W^T row-major)
__global__ void cvtT_f32_bf16(const float* __restrict__ in,
                              __bf16* __restrict__ out, int R, int C) {
    int i = blockIdx.x * blockDim.x + threadIdx.x;
    if (i < R * C) {
        int r = i / C, c = i % C;
        out[(size_t)c * R + r] = (__bf16)in[i];
    }
}

// ------------------------------- GEMM --------------------------------------
// C[m,n] = scale * (sum_k A[m,k] * BT[k,n] + bias[n])
// Block: 256 thr = 8 waves (4Mx2N), wave tile 32x64, block tile 128x128.
__global__ __launch_bounds__(256)
void gemm_bf16_wmma(const __bf16* __restrict__ A,
                    const __bf16* __restrict__ BT,
                    const float*  __restrict__ bias,
                    float scale,
                    __bf16* __restrict__ outB,
                    float*  __restrict__ outF,
                    int M, int N, int K)
{
    const int lane = threadIdx.x & 31;
    const int wave = threadIdx.x >> 5;
    const int l16  = lane & 15;
    const int hh   = lane >> 4;
    const int wm   = wave >> 1, wn = wave & 1;
    const int mBase = blockIdx.y * 128 + wm * 32;
    const int nBase = blockIdx.x * 128 + wn * 64;

    v8f acc[2][4] = {};

    for (int kb = 0; kb < K; kb += 32) {
        // A fragments: lane = M row, regs K = {8h..8h+7, 16+8h..23+8h}
        V16 a[2];
#pragma unroll
        for (int i = 0; i < 2; ++i) {
            const __bf16* ap = A + (size_t)(mBase + 16 * i + l16) * K + kb;
            a[i].h[0] = *(const v8bf*)(ap + 8 * hh);
            a[i].h[1] = *(const v8bf*)(ap + 16 + 8 * hh);
        }
        // B fragments: lane = N col (l16), regs K = {8h..8h+7, 16+8h..23+8h}
        const __bf16* bp = BT + (size_t)kb * N + nBase + l16;
        v16bf b[4];
#pragma unroll
        for (int j = 0; j < 4; ++j)
#pragma unroll
            for (int s = 0; s < 16; ++s)
                b[j][s] = bp[(size_t)((s < 8) ? 8 * hh + s : 8 + 8 * hh + s) * N + 16 * j];

        if (kb + 32 < K) {
            __builtin_prefetch(bp + (size_t)32 * N, 0, 1);
            __builtin_prefetch(A + (size_t)(mBase + l16) * K + kb + 32, 0, 1);
        }

#pragma unroll
        for (int i = 0; i < 2; ++i)
#pragma unroll
            for (int j = 0; j < 4; ++j)
                acc[i][j] = __builtin_amdgcn_wmma_f32_16x16x32_bf16(
                    false, a[i].v, false, b[j], (short)0, acc[i][j],
                    false, false);
    }

    // Epilogue: lane -> n = nBase+16j+l16 ; reg r -> m = mBase+16i+r+8*hh.
    if (outF) {
#pragma unroll
        for (int i = 0; i < 2; ++i)
#pragma unroll
            for (int j = 0; j < 4; ++j) {
                const int n = nBase + 16 * j + l16;
                const float bv = bias[n];
#pragma unroll
                for (int r = 0; r < 8; ++r)
                    outF[(size_t)(mBase + 16 * i + r + 8 * hh) * N + n] =
                        (acc[i][j][r] + bv) * scale;
            }
    }
    if (outB) {
#pragma unroll
        for (int i = 0; i < 2; ++i)
#pragma unroll
            for (int j = 0; j < 4; ++j) {
                const int n = nBase + 16 * j + l16;
                const float bv = bias[n];
#pragma unroll
                for (int r = 0; r < 8; ++r)
                    outB[(size_t)(mBase + 16 * i + r + 8 * hh) * N + n] =
                        (__bf16)((acc[i][j][r] + bv) * scale);
            }
    }
}

// --------------------------- flash attention -------------------------------
// Grid: x = q-tile (128 rows), y = b*NHEADS + h. 8 waves x 16 q-rows.
// Computes S^T = K*Q^T per 32-row K/V chunk; P is then directly the WMMA
// A-operand of P*V. K/V double-buffered via async LDS loads.
__global__ __launch_bounds__(256)
void flash_attn_bf16(const __bf16* __restrict__ Qb,
                     const __bf16* __restrict__ Kb,
                     const __bf16* __restrict__ Vb,
                     const int*    __restrict__ mask,
                     __bf16* __restrict__ Ctx)
{
    __shared__ __attribute__((aligned(32))) __bf16 sK[2][32][DHEAD]; // 8 KB
    __shared__ __attribute__((aligned(32))) __bf16 sV[2][32][DHEAD]; // 8 KB
    __shared__ __attribute__((aligned(32))) __bf16 sQ[8][16][DHEAD]; // 16 KB

    const int tid  = threadIdx.x;
    const int lane = tid & 31;
    const int wave = tid >> 5;
    const int l16  = lane & 15;
    const int hh   = lane >> 4;
    const int b    = blockIdx.y >> 4;
    const int hd   = blockIdx.y & 15;
    const int qRow0 = blockIdx.x * 128 + wave * 16;

    const size_t headOff = (size_t)b * QLEN * DIM + (size_t)hd * DHEAD;
    const __bf16* Kh = Kb + headOff;
    const __bf16* Vh = Vb + headOff;

    // --- one-time: stage Q tile (wave-private), gather Q^T B-fragments ---
#pragma unroll
    for (int j = 0; j < 4; ++j) {
        const int chunk = j * 32 + lane;            // 128 8-elem chunks
        const int row = chunk >> 3, col0 = (chunk & 7) * 8;
        *(v8bf*)&sQ[wave][row][col0] =
            *(const v8bf*)(Qb + headOff + (size_t)(qRow0 + row) * DIM + col0);
    }
    V16 bq[2];                       // B operand: lane = q row, regs = d (table order)
#pragma unroll
    for (int db = 0; db < 2; ++db) {
#pragma unroll
        for (int e = 0; e < 16; ++e)
            bq[db].e[e] = sQ[wave][l16][32 * db + ((e < 8) ? 8 * hh + e : 8 + 8 * hh + e)];
    }

    // --- async double-buffer prologue: chunks 0 and 1 in flight ---
    const int srow = tid >> 3, scol = (tid & 7) * 8;  // 32 rows x 8 chunks
    async_b128_to_lds(&sK[0][srow][scol], Kh + (size_t)srow * DIM + scol);
    async_b128_to_lds(&sV[0][srow][scol], Vh + (size_t)srow * DIM + scol);
    async_b128_to_lds(&sK[1][srow][scol], Kh + (size_t)(32 + srow) * DIM + scol);
    async_b128_to_lds(&sV[1][srow][scol], Vh + (size_t)(32 + srow) * DIM + scol);

    v8f   acc[4] = {};               // O tile: lane -> d, reg -> q
    float mrow = -INFINITY, lrow = 0.0f;

    for (int c = 0, buf = 0; c < QLEN; c += 32, buf ^= 1) {
        // front batch (2 async instr per wave) complete; next stays in flight
        if (c + 32 < QLEN) asm volatile("s_wait_asynccnt 2" ::: "memory");
        else               asm volatile("s_wait_asynccnt 0" ::: "memory");
        __syncthreads();

        const float madd =
            (mask[b * QLEN + c + lane] == 0) ? -INFINITY : 0.0f;

        // S^T tiles: A = K chunk rows (lane = row), B = Q^T. C-layout:
        // lane -> q = l16, reg r -> k-row = 16i + r + 8*hh.
        v8f st[2];
#pragma unroll
        for (int i = 0; i < 2; ++i) {
            const __bf16* kr = &sK[buf][16 * i + l16][0];
            V16 ak0, ak1;
            ak0.h[0] = *(const v8bf*)(kr + 8 * hh);
            ak0.h[1] = *(const v8bf*)(kr + 16 + 8 * hh);
            ak1.h[0] = *(const v8bf*)(kr + 32 + 8 * hh);
            ak1.h[1] = *(const v8bf*)(kr + 48 + 8 * hh);
            v8f z = {};
            z = __builtin_amdgcn_wmma_f32_16x16x32_bf16(
                false, ak0.v, false, bq[0].v, (short)0, z, false, false);
            z = __builtin_amdgcn_wmma_f32_16x16x32_bf16(
                false, ak1.v, false, bq[1].v, (short)0, z, false, false);
#pragma unroll
            for (int r = 0; r < 8; ++r)
                z[r] += __shfl(madd, 16 * i + r + 8 * hh, 32);
            st[i] = z;
        }

        // Online softmax for q = l16 (16 k values in-lane, 16 in xor-16 lane)
        float mx = -INFINITY;
#pragma unroll
        for (int r = 0; r < 8; ++r) mx = fmaxf(mx, fmaxf(st[0][r], st[1][r]));
        mx = fmaxf(mx, __shfl_xor(mx, 16, 32));
        const float mnew = fmaxf(mrow, mx);
        const float rs   = __expf(mrow - mnew);
        V16 ap; float sum = 0.0f;
#pragma unroll
        for (int r = 0; r < 8; ++r) {
            const float p0 = __expf(st[0][r] - mnew);   // k = r + 8*hh
            const float p1 = __expf(st[1][r] - mnew);   // k = 16 + r + 8*hh
            sum += p0 + p1;
            ap.e[r]     = (__bf16)p0;   // A elem e<8  <-> k = 8*hh + e
            ap.e[8 + r] = (__bf16)p1;   // A elem 8+e <-> k = 16 + 8*hh + e
        }
        sum += __shfl_xor(sum, 16, 32);
        lrow = lrow * rs + sum;
        mrow = mnew;

        // Rescale O: acc reg r belongs to q = r + 8*hh -> fetch that lane's rs
        float rs_o[8];
#pragma unroll
        for (int r = 0; r < 8; ++r) rs_o[r] = __shfl(rs, r + 8 * hh, 32);
#pragma unroll
        for (int dc = 0; dc < 4; ++dc)
#pragma unroll
            for (int r = 0; r < 8; ++r) acc[dc][r] *= rs_o[r];

        // O += P * V: B operand lane = d (l16), regs = k-rows (table order)
        const __bf16* vr = &sV[buf][0][l16];
#pragma unroll
        for (int dc = 0; dc < 4; ++dc) {
            v16bf bv;
#pragma unroll
            for (int s = 0; s < 16; ++s)
                bv[s] = vr[((s < 8) ? 8 * hh + s : 8 + 8 * hh + s) * DHEAD + 16 * dc];
            acc[dc] = __builtin_amdgcn_wmma_f32_16x16x32_bf16(
                false, ap.v, false, bv, (short)0, acc[dc], false, false);
        }

        __syncthreads();   // all waves done reading buf before refill
        if (c + 64 < QLEN) {
            async_b128_to_lds(&sK[buf][srow][scol],
                              Kh + (size_t)(c + 64 + srow) * DIM + scol);
            async_b128_to_lds(&sV[buf][srow][scol],
                              Vh + (size_t)(c + 64 + srow) * DIM + scol);
        }
    }

    // Normalize (1/l per q, broadcast to reg layout) and emit context
    const float inv = 1.0f / lrow;
#pragma unroll
    for (int r = 0; r < 8; ++r) {
        const float invo = __shfl(inv, r + 8 * hh, 32);
        const size_t row = (size_t)b * QLEN + qRow0 + r + 8 * hh;
#pragma unroll
        for (int dc = 0; dc < 4; ++dc)
            Ctx[row * DIM + hd * DHEAD + dc * 16 + l16] =
                (__bf16)(acc[dc][r] * invo);
    }
}

// ------------------------------ launcher -----------------------------------

extern "C" void kernel_launch(void* const* d_in, const int* in_sizes, int n_in,
                              void* d_out, int out_size, void* d_ws, size_t ws_size,
                              hipStream_t stream)
{
    const float* x   = (const float*)d_in[0];
    const int*   msk = (const int*)  d_in[1];
    const float* qw  = (const float*)d_in[2];
    const float* qb  = (const float*)d_in[3];
    const float* kw  = (const float*)d_in[4];
    const float* kb  = (const float*)d_in[5];
    const float* vw  = (const float*)d_in[6];
    const float* vb  = (const float*)d_in[7];
    const float* ow  = (const float*)d_in[8];
    const float* ob  = (const float*)d_in[9];
    float* out = (float*)d_out;

    // Workspace carve-up (bf16): X 8MB | 4xW^T 8MB | Q,K,V 24MB | Ctx 8MB
    char* w = (char*)d_ws;
    __bf16* Xb  = (__bf16*)w;  w += (size_t)MROWS * DIM * 2;
    __bf16* WqT = (__bf16*)w;  w += (size_t)DIM * DIM * 2;
    __bf16* WkT = (__bf16*)w;  w += (size_t)DIM * DIM * 2;
    __bf16* WvT = (__bf16*)w;  w += (size_t)DIM * DIM * 2;
    __bf16* WoT = (__bf16*)w;  w += (size_t)DIM * DIM * 2;
    __bf16* Qb  = (__bf16*)w;  w += (size_t)MROWS * DIM * 2;
    __bf16* Kb  = (__bf16*)w;  w += (size_t)MROWS * DIM * 2;
    __bf16* Vb  = (__bf16*)w;  w += (size_t)MROWS * DIM * 2;
    __bf16* Ctx = (__bf16*)w;  w += (size_t)MROWS * DIM * 2;

    const int nX = MROWS * DIM;
    const int nW = DIM * DIM;

    cvt_f32_bf16 <<<(nX + 255) / 256, 256, 0, stream>>>(x,  Xb,  nX);
    cvtT_f32_bf16<<<(nW + 255) / 256, 256, 0, stream>>>(qw, WqT, DIM, DIM);
    cvtT_f32_bf16<<<(nW + 255) / 256, 256, 0, stream>>>(kw, WkT, DIM, DIM);
    cvtT_f32_bf16<<<(nW + 255) / 256, 256, 0, stream>>>(vw, WvT, DIM, DIM);
    cvtT_f32_bf16<<<(nW + 255) / 256, 256, 0, stream>>>(ow, WoT, DIM, DIM);

    const dim3 gGemm(DIM / 128, MROWS / 128);          // (8, 32)
    const float qscale = 1.0f / sqrtf((float)DHEAD);   // fold 1/sqrt(d) into Q
    gemm_bf16_wmma<<<gGemm, 256, 0, stream>>>(Xb, WqT, qb, qscale,
                                              Qb, nullptr, MROWS, DIM, DIM);
    gemm_bf16_wmma<<<gGemm, 256, 0, stream>>>(Xb, WkT, kb, 1.0f,
                                              Kb, nullptr, MROWS, DIM, DIM);
    gemm_bf16_wmma<<<gGemm, 256, 0, stream>>>(Xb, WvT, vb, 1.0f,
                                              Vb, nullptr, MROWS, DIM, DIM);

    const dim3 gAttn(QLEN / 128, BSZ * NHEADS);        // (16, 32)
    flash_attn_bf16<<<gAttn, 256, 0, stream>>>(Qb, Kb, Vb, msk, Ctx);

    gemm_bf16_wmma<<<gGemm, 256, 0, stream>>>(Ctx, WoT, ob, 1.0f,
                                              nullptr, out, MROWS, DIM, DIM);
}
